// HViT_22239340659139
// MI455X (gfx1250) — compile-verified, matrix operand order checked by reading, not run on hardware
//
#include <hip/hip_runtime.h>

#define D_   768
#define LTOK 196
#define ROWS 6272   // 32*196
#define HEPS 1e-6f
#define WSZ  (768*768)

typedef __attribute__((ext_vector_type(16))) __bf16 v16bf;
typedef __attribute__((ext_vector_type(8)))  __bf16 bf16x8;
typedef __attribute__((ext_vector_type(8)))  float  v8f;

__device__ __forceinline__ __bf16 to_bf(float x){ return (__bf16)x; }

// Build a 16-elem WMMA A/B fragment: elements 0..7 = K at p[0..7],
// elements 8..15 = K at p[16..23] (caller adds lane-half*8 to base).
// Works on global bf16 (two global_load_b128) or LDS bf16 (two ds_load_b128).
__device__ __forceinline__ v16bf ldfrag(const __bf16* p){
  bf16x8 lo = *(const bf16x8*)p;
  bf16x8 hi = *(const bf16x8*)(p + 16);
  return __builtin_shufflevector(lo, hi, 0,1,2,3,4,5,6,7,8,9,10,11,12,13,14,15);
}

__device__ __forceinline__ v8f wmma_bf16(v16bf a, v16bf b, v8f c){
  return __builtin_amdgcn_wmma_f32_16x16x32_bf16(false, a, false, b, (short)0, c, false, false);
}

__device__ __forceinline__ float block_sum256(float v, float* sb){
  #pragma unroll
  for (int off = 16; off; off >>= 1) v += __shfl_xor(v, off, 32);
  __syncthreads();
  if ((threadIdx.x & 31) == 0) sb[threadIdx.x >> 5] = v;
  __syncthreads();
  float tot = 0.f;
  #pragma unroll
  for (int i = 0; i < 8; i++) tot += sb[i];
  return tot;
}

// ---------------------------------------------------------------------------
// One-time weight transpose+convert: W[k][n] f32 -> Wt[n][k] bf16
// ---------------------------------------------------------------------------
__global__ __launch_bounds__(256)
void wtrans_k(const float* __restrict__ W, __bf16* __restrict__ Wt)
{
  __shared__ float t[64][65];
  const int tid = threadIdx.x;
  const int k0 = blockIdx.y * 64, n0 = blockIdx.x * 64;
  for (int i = tid; i < 64 * 64; i += 256){
    int r = i >> 6, c = i & 63;
    t[r][c] = W[(size_t)(k0 + r) * D_ + n0 + c];
  }
  __syncthreads();
  for (int i = tid; i < 64 * 64; i += 256){
    int nr = i >> 6, kc = i & 63;
    Wt[(size_t)(n0 + nr) * D_ + k0 + kc] = to_bf(t[kc][nr]);
  }
}

// ---------------------------------------------------------------------------
// LDS-free GEMM: C[M,N] = A[M,K](bf16,[m][k]) @ Bt[N,K](bf16,[n][k])
// Fragments loaded directly from global (L2/WGP$-resident operands).
// Block = 128 threads = 4 waves (2x2); block tile 128x128; wave tile 64x64.
// 16 WMMAs per 16 b128 loads per 32-K step (each fragment reused 4x).
// ---------------------------------------------------------------------------
enum { EP_BIAS = 1, EP_RELU = 2, EP_POS = 4, EP_F32 = 8, EP_BF16 = 16 };

__global__ __launch_bounds__(128)
void gemm_l2_k(const __bf16* __restrict__ A, const __bf16* __restrict__ Bt,
               const float* __restrict__ bias, const float* __restrict__ pos,
               float* __restrict__ Cf, __bf16* __restrict__ Cb,
               int N, int K, int flags)
{
  const int tid = threadIdx.x;
  const int w = tid >> 5, lane = tid & 31, lh = lane >> 4, l15 = lane & 15;
  const int wr = w >> 1, wc = w & 1;
  const int gm = blockIdx.y * 128 + wr * 64;
  const int gn = blockIdx.x * 128 + wc * 64;

  v8f zero = {0.f,0.f,0.f,0.f,0.f,0.f,0.f,0.f};
  v8f acc[4][4];
  #pragma unroll
  for (int mi = 0; mi < 4; mi++)
    #pragma unroll
    for (int j = 0; j < 4; j++) acc[mi][j] = zero;

  const __bf16* a0 = A + (size_t)(gm + l15) * K;
  const __bf16* b0 = Bt + (size_t)(gn + l15) * K;

  for (int k0 = 0; k0 < K; k0 += 32){
    const int ko = k0 + lh * 8;
    v16bf af[4];
    #pragma unroll
    for (int mi = 0; mi < 4; mi++)
      af[mi] = ldfrag(a0 + (size_t)(mi * 16) * K + ko);
    #pragma unroll
    for (int j = 0; j < 4; j++){
      v16bf bfv = ldfrag(b0 + (size_t)(j * 16) * K + ko);
      #pragma unroll
      for (int mi = 0; mi < 4; mi++)
        acc[mi][j] = wmma_bf16(af[mi], bfv, acc[mi][j]);
    }
  }

  #pragma unroll
  for (int mi = 0; mi < 4; mi++){
    #pragma unroll
    for (int j = 0; j < 4; j++){
      #pragma unroll
      for (int r = 0; r < 8; r++){
        int row = gm + mi * 16 + r + lh * 8;
        int col = gn + j * 16 + l15;
        float v = acc[mi][j][r];
        if (flags & EP_BIAS) v += bias[col];
        if (flags & EP_POS)  v += pos[(size_t)(row % LTOK) * N + col];
        if (flags & EP_RELU) v = fmaxf(v, 0.f);
        if (flags & EP_F32)  Cf[(size_t)row * N + col] = v;
        if (flags & EP_BF16) Cb[(size_t)row * N + col] = to_bf(v);
      }
    }
  }
}

// ---------------------------------------------------------------------------
// Hyperbolic attention, one workgroup per (qblock, head, batch).
// Q/K/V are bf16 [row][768]; output o is bf16 (only ever a GEMM operand).
// Keys padded 196->224, dh padded 64->80 (col 64 = v time component).
// ---------------------------------------------------------------------------
#define ALK  224
#define ADH  64
#define ADHP 80
#define AMQ  64
#define KSTR 72
#define VSTR 232
#define ATT_SMEM ((ALK*KSTR + ADHP*VSTR + AMQ*KSTR + AMQ*VSTR)*2 + (ALK + AMQ)*4)

__global__ __launch_bounds__(128)
void attn_k(const __bf16* __restrict__ q, const __bf16* __restrict__ kk,
            const __bf16* __restrict__ vv, __bf16* __restrict__ ob)
{
  extern __shared__ __align__(16) char smem_raw[];
  __bf16* Ks = (__bf16*)smem_raw;            // [ALK][KSTR]  key spatial
  __bf16* Vt = Ks + ALK * KSTR;              // [ADHP][VSTR] V^T (+time row 64)
  __bf16* Qs = Vt + ADHP * VSTR;             // [AMQ][KSTR]
  __bf16* Pr = Qs + AMQ * KSTR;              // [AMQ][VSTR]  softmax probs
  float* ktime = (float*)(Pr + AMQ * VSTR);  // [ALK]
  float* qtime = ktime + ALK;                // [AMQ]

  const int tid = threadIdx.x;
  const int b = blockIdx.z, h = blockIdx.y, qb = blockIdx.x;
  const int w = tid >> 5, lane = tid & 31, lh = lane >> 4, l15 = lane & 15;
  const size_t base = (size_t)b * LTOK * D_ + (size_t)h * ADH;
  const int q0 = qb * AMQ;

  bf16x8 zc;
  #pragma unroll
  for (int i = 0; i < 8; i++) zc[i] = (__bf16)0.f;

  // K: straight bf16 b128 copies (8 chunks of 8 per token row)
  for (int idx = tid; idx < ALK * 8; idx += 128){
    int t = idx >> 3, c8 = (idx & 7) << 3;
    bf16x8 ch = (t < LTOK) ? *(const bf16x8*)(kk + base + (size_t)t * D_ + c8) : zc;
    *(bf16x8*)&Ks[t * KSTR + c8] = ch;
  }
  // V: transpose scatter into Vt[d][t]
  for (int idx = tid; idx < ALK * 8; idx += 128){
    int t = idx >> 3, c8 = (idx & 7) << 3;
    bf16x8 ch = (t < LTOK) ? *(const bf16x8*)(vv + base + (size_t)t * D_ + c8) : zc;
    #pragma unroll
    for (int j = 0; j < 8; j++) Vt[(c8 + j) * VSTR + t] = ch[j];
  }
  for (int i = tid; i < 16 * VSTR; i += 128)   // zero rows 64..79
    Vt[(size_t)64 * VSTR + i] = (__bf16)0.f;
  // Q
  for (int idx = tid; idx < AMQ * 8; idx += 128){
    int r = idx >> 3, c8 = (idx & 7) << 3;
    int t = q0 + r;
    bf16x8 ch = (t < LTOK) ? *(const bf16x8*)(q + base + (size_t)t * D_ + c8) : zc;
    *(bf16x8*)&Qs[r * KSTR + c8] = ch;
  }
  __syncthreads();
  for (int t = tid; t < ALK; t += 128){
    float s = 0.f, sv = 0.f;
    for (int d2 = 0; d2 < ADH; ++d2){
      float a = (float)Ks[t * KSTR + d2];  s  += a * a;
      float c = (float)Vt[d2 * VSTR + t];  sv += c * c;
    }
    ktime[t] = sqrtf(1.f + s);
    Vt[64 * VSTR + t] = to_bf(sqrtf(1.f + sv));   // v time component row
  }
  for (int r = tid; r < AMQ; r += 128){
    float s = 0.f;
    for (int d2 = 0; d2 < ADH; ++d2){
      float a = (float)Qs[r * KSTR + d2]; s += a * a;
    }
    qtime[r] = sqrtf(1.f + s);
  }
  __syncthreads();

  v8f zero = {0.f,0.f,0.f,0.f,0.f,0.f,0.f,0.f};
  v8f acc[14];
  #pragma unroll
  for (int j = 0; j < 14; j++) acc[j] = zero;

  // logits = q_s . k_s  (Lorentz correction applied below)
  #pragma unroll
  for (int kc = 0; kc < 2; ++kc){
    int k0 = kc * 32 + lh * 8;
    v16bf af = ldfrag(&Qs[(w * 16 + l15) * KSTR + k0]);
    #pragma unroll
    for (int jt = 0; jt < 14; ++jt){
      v16bf bfv = ldfrag(&Ks[(jt * 16 + l15) * KSTR + k0]);
      acc[jt] = wmma_bf16(af, bfv, acc[jt]);
    }
  }

  // masked softmax per query row (row lives in one 16-lane group)
  #pragma unroll
  for (int r = 0; r < 8; ++r){
    int mloc = w * 16 + r + lh * 8;
    float qt = qtime[mloc];
    float vals[14];
    float mx = -1e30f;
    #pragma unroll
    for (int jt = 0; jt < 14; ++jt){
      int n = jt * 16 + l15;
      float lg = 0.25f + 0.25f * (acc[jt][r] - qt * ktime[n]); // (2+2*inner)/8
      if (n >= LTOK) lg = -1e30f;
      vals[jt] = lg;
      mx = fmaxf(mx, lg);
    }
    #pragma unroll
    for (int off = 1; off < 16; off <<= 1) mx = fmaxf(mx, __shfl_xor(mx, off, 32));
    float sum = 0.f;
    #pragma unroll
    for (int jt = 0; jt < 14; ++jt){
      float e = __expf(vals[jt] - mx); vals[jt] = e; sum += e;
    }
    #pragma unroll
    for (int off = 1; off < 16; off <<= 1) sum += __shfl_xor(sum, off, 32);
    float inv = 1.f / sum;
    #pragma unroll
    for (int jt = 0; jt < 14; ++jt)
      Pr[mloc * VSTR + jt * 16 + l15] = to_bf(vals[jt] * inv);
  }

  // s = a @ [v_s | v_t]
  v8f acc2[5];
  #pragma unroll
  for (int j = 0; j < 5; j++) acc2[j] = zero;
  #pragma unroll
  for (int kc = 0; kc < 7; ++kc){
    int k0 = kc * 32 + lh * 8;
    v16bf af = ldfrag(&Pr[(w * 16 + l15) * VSTR + k0]);
    #pragma unroll
    for (int nt = 0; nt < 5; ++nt){
      v16bf bfv = ldfrag(&Vt[(nt * 16 + l15) * VSTR + k0]);
      acc2[nt] = wmma_bf16(af, bfv, acc2[nt]);
    }
  }

  // Lorentz normalize s and store spatial part (bf16, GEMM operand layout)
  #pragma unroll
  for (int r = 0; r < 8; ++r){
    int mloc = w * 16 + r + lh * 8;
    int grow = q0 + mloc;
    float sv0 = acc2[0][r], sv1 = acc2[1][r], sv2 = acc2[2][r], sv3 = acc2[3][r];
    float ss = sv0*sv0 + sv1*sv1 + sv2*sv2 + sv3*sv3;
    #pragma unroll
    for (int off = 1; off < 16; off <<= 1) ss += __shfl_xor(ss, off, 32);
    float st  = __shfl(acc2[4][r], lh * 16, 32);   // col 64 = time, lane l15==0
    float inv = 1.f / sqrtf(fmaxf(st * st - ss, HEPS));
    if (grow < LTOK){
      __bf16* op = ob + ((size_t)b * LTOK + grow) * D_ + (size_t)h * ADH + l15;
      op[0]  = to_bf(sv0 * inv);
      op[16] = to_bf(sv1 * inv);
      op[32] = to_bf(sv2 * inv);
      op[48] = to_bf(sv3 * inv);
    }
  }
}

// ---------------------------------------------------------------------------
// Elementwise / reduction kernels
// ---------------------------------------------------------------------------
__global__ __launch_bounds__(256)
void patchify_k(const float* __restrict__ x, __bf16* __restrict__ outb)
{
  int row = blockIdx.x;
  int b = row / LTOK, n = row % LTOK;
  int py = n / 14, px = n % 14;
  for (int e = threadIdx.x; e < D_; e += 256){
    int c = e % 3;
    int j = (e / 3) % 16;
    int i = e / 48;
    outb[(size_t)row * D_ + e] =
      to_bf(x[(((size_t)b * 3 + c) * 224 + py * 16 + i) * 224 + px * 16 + j]);
  }
}

__global__ __launch_bounds__(256)
void expmap0_k(const float* __restrict__ u, float* __restrict__ out,
               __bf16* __restrict__ outh)
{
  __shared__ float sb[8];
  size_t row = blockIdx.x;
  float uv[3]; float ss = 0.f;
  #pragma unroll
  for (int i = 0; i < 3; i++){
    int d = threadIdx.x + i * 256;
    uv[i] = u[row * D_ + d];
    ss += uv[i] * uv[i];
  }
  ss = block_sum256(ss, sb);
  float nn = fminf(fmaxf(sqrtf(ss), HEPS), 10.f);
  float sc = sinhf(nn) / nn;
  #pragma unroll
  for (int i = 0; i < 3; i++){
    int d = threadIdx.x + i * 256;
    float v = uv[i] * sc;
    out[row * D_ + d] = v;
    outh[row * D_ + d] = to_bf(v);
  }
}

// out = hlayernorm( lorentz_normalize(z + x) , g, b) -- spatial parts only
__global__ __launch_bounds__(256)
void res_norm_k(const float* __restrict__ z, const float* __restrict__ x,
                const float* __restrict__ g, const float* __restrict__ bb,
                float* __restrict__ out, __bf16* __restrict__ outh)
{
  __shared__ float sb[8];
  size_t row = blockIdx.x;
  float zv[3], xv[3];
  float sz = 0.f, sx = 0.f, sd = 0.f, ssum = 0.f;
  #pragma unroll
  for (int i = 0; i < 3; i++){
    int d = threadIdx.x + i * 256;
    float a = z[row * D_ + d], c = x[row * D_ + d];
    zv[i] = a; xv[i] = c;
    sz += a * a; sx += c * c; sd += a * c; ssum += a + c;
  }
  sz   = block_sum256(sz, sb);
  sx   = block_sum256(sx, sb);
  sd   = block_sum256(sd, sb);
  ssum = block_sum256(ssum, sb);
  float st  = sqrtf(1.f + sz) + sqrtf(1.f + sx);
  float s2  = sz + sx + 2.f * sd;
  float inv = 1.f / sqrtf(fmaxf(st * st - s2, HEPS));
  float mu  = inv * ssum * (1.f / 768.f);
  float var = inv * inv * s2 * (1.f / 768.f) - mu * mu;
  float rs  = rsqrtf(var + 1e-5f);
  #pragma unroll
  for (int i = 0; i < 3; i++){
    int d = threadIdx.x + i * 256;
    float v = ((zv[i] + xv[i]) * inv - mu) * rs * g[d] + bb[d];
    out[row * D_ + d] = v;
    outh[row * D_ + d] = to_bf(v);
  }
}

__global__ __launch_bounds__(256)
void tnorm_k(const float* __restrict__ hb, float* __restrict__ tn)
{
  __shared__ float sb[8];
  size_t row = blockIdx.x;
  float ss = 0.f;
  #pragma unroll
  for (int i = 0; i < 3; i++){
    int d = threadIdx.x + i * 256;
    float v = hb[row * D_ + d];
    ss += v * v;
  }
  ss = block_sum256(ss, sb);
  if (threadIdx.x == 0) tn[row] = sqrtf(1.f + ss);
}

__global__ __launch_bounds__(256)
void pool_k(const float* __restrict__ hb, const float* __restrict__ tn,
            float* __restrict__ ps, float* __restrict__ pt)
{
  __shared__ float sb[8];
  int b = blockIdx.x;
  float st = 0.f;
  for (int n = threadIdx.x; n < LTOK; n += 256) st += tn[b * LTOK + n];
  st = block_sum256(st, sb);
  float sums[3]; float ssq = 0.f;
  #pragma unroll
  for (int i = 0; i < 3; i++){
    int d = threadIdx.x + i * 256;
    float s = 0.f;
    for (int n = 0; n < LTOK; n++) s += hb[((size_t)b * LTOK + n) * D_ + d];
    sums[i] = s; ssq += s * s;
  }
  ssq = block_sum256(ssq, sb);
  float inv = 1.f / sqrtf(fmaxf(st * st - ssq, HEPS));
  #pragma unroll
  for (int i = 0; i < 3; i++){
    int d = threadIdx.x + i * 256;
    ps[b * D_ + d] = sums[i] * inv;
  }
  if (threadIdx.x == 0) pt[b] = st * inv;
}

__global__ __launch_bounds__(256)
void cls_k(const float* __restrict__ ps, const float* __restrict__ pt,
           const float* __restrict__ Wc, const float* __restrict__ bc,
           float* __restrict__ out)
{
  int b = blockIdx.x;
  int c = blockIdx.y * 256 + threadIdx.x;
  if (c >= 1000) return;
  float acc = 0.f, wsq = 0.f;
  for (int d = 0; d < D_; ++d){
    float wv = Wc[(size_t)c * D_ + d];
    acc += ps[b * D_ + d] * wv;
    wsq += wv * wv;
  }
  out[b * 1000 + c] = 2.f + 2.f * (acc - pt[b] * sqrtf(1.f + wsq)) + bc[c];
}

// ---------------------------------------------------------------------------
// Host orchestration
// ---------------------------------------------------------------------------
extern "C" void kernel_launch(void* const* d_in, const int* in_sizes, int n_in,
                              void* d_out, int out_size, void* d_ws, size_t ws_size,
                              hipStream_t stream)
{
  (void)in_sizes; (void)n_in; (void)out_size; (void)ws_size;
  const float* x   = (const float*)d_in[0];
  const float* Wp  = (const float*)d_in[1];
  const float* bp  = (const float*)d_in[2];
  const float* pos = (const float*)d_in[3];
  const float* Wc  = (const float*)d_in[196];
  const float* bc  = (const float*)d_in[197];
  auto blk = (+[](void* const* din, int l, int j){
    return (const float*)din[4 + l * 16 + j];
  });

  const size_t SZ = (size_t)ROWS * D_;
  char* p = (char*)d_ws;
  float*  bX   = (float*)(p + 0 * SZ * 4);
  float*  bX1  = (float*)(p + 1 * SZ * 4);
  float*  bZ   = (float*)(p + 2 * SZ * 4);
  __bf16* bXh  = (__bf16*)(p + 3 * SZ * 4);
  __bf16* bX1h = bXh + 1 * SZ;
  __bf16* bHh  = bXh + 2 * SZ;
  __bf16* bOh  = bXh + 3 * SZ;
  __bf16* bQh  = bXh + 4 * SZ;
  __bf16* bKh  = bXh + 5 * SZ;
  __bf16* bVh  = bXh + 6 * SZ;
  __bf16* bPh  = bXh + 7 * SZ;
  __bf16* wTp  = bXh + 8 * SZ;            // Wp^T bf16
  __bf16* wT0  = wTp + WSZ;               // 6 per-layer weight slots (reused)
  float*  tbuf = (float*)((char*)(wT0 + 6 * (size_t)WSZ));
  float*  pooled_s = tbuf + ROWS;
  float*  pooled_t = pooled_s + 32 * D_;

  dim3 wtg(12, 12), wtb(256);
  dim3 gg(D_ / 128, ROWS / 128), gb(128);   // (6, 49)

  // patch embed + pos, then expmap0
  wtrans_k<<<wtg, wtb, 0, stream>>>(Wp, wTp);
  patchify_k<<<ROWS, 256, 0, stream>>>(x, bPh);
  gemm_l2_k<<<gg, gb, 0, stream>>>(bPh, wTp, bp, pos, bZ, nullptr,
                                   D_, D_, EP_BIAS | EP_POS | EP_F32);
  expmap0_k<<<ROWS, 256, 0, stream>>>(bZ, bX, bXh);

  for (int l = 0; l < 12; ++l){
    const float* Wq = blk(d_in,l,0);  const float* bq = blk(d_in,l,1);
    const float* Wk = blk(d_in,l,2);  const float* bk = blk(d_in,l,3);
    const float* Wv = blk(d_in,l,4);  const float* bv = blk(d_in,l,5);
    const float* Wo = blk(d_in,l,6);  const float* bo = blk(d_in,l,7);
    const float* W1 = blk(d_in,l,8);  const float* bm1 = blk(d_in,l,9);
    const float* W2 = blk(d_in,l,10); const float* bm2 = blk(d_in,l,11);
    const float* g1 = blk(d_in,l,12); const float* b1 = blk(d_in,l,13);
    const float* g2 = blk(d_in,l,14); const float* b2 = blk(d_in,l,15);

    // per-layer weight slots (stream order serializes reuse)
    wtrans_k<<<wtg, wtb, 0, stream>>>(Wq, wT0 + 0 * (size_t)WSZ);
    wtrans_k<<<wtg, wtb, 0, stream>>>(Wk, wT0 + 1 * (size_t)WSZ);
    wtrans_k<<<wtg, wtb, 0, stream>>>(Wv, wT0 + 2 * (size_t)WSZ);
    wtrans_k<<<wtg, wtb, 0, stream>>>(Wo, wT0 + 3 * (size_t)WSZ);
    wtrans_k<<<wtg, wtb, 0, stream>>>(W1, wT0 + 4 * (size_t)WSZ);
    wtrans_k<<<wtg, wtb, 0, stream>>>(W2, wT0 + 5 * (size_t)WSZ);

    gemm_l2_k<<<gg, gb, 0, stream>>>(bXh, wT0 + 0 * (size_t)WSZ, bq, nullptr,
                                     nullptr, bQh, D_, D_, EP_BIAS | EP_BF16);
    gemm_l2_k<<<gg, gb, 0, stream>>>(bXh, wT0 + 1 * (size_t)WSZ, bk, nullptr,
                                     nullptr, bKh, D_, D_, EP_BIAS | EP_BF16);
    gemm_l2_k<<<gg, gb, 0, stream>>>(bXh, wT0 + 2 * (size_t)WSZ, bv, nullptr,
                                     nullptr, bVh, D_, D_, EP_BIAS | EP_BF16);
    attn_k<<<dim3(4, 12, 32), 128, ATT_SMEM, stream>>>(bQh, bKh, bVh, bOh);
    gemm_l2_k<<<gg, gb, 0, stream>>>(bOh, wT0 + 3 * (size_t)WSZ, bo, nullptr,
                                     bZ, nullptr, D_, D_, EP_BIAS | EP_F32);
    res_norm_k<<<ROWS, 256, 0, stream>>>(bZ, bX, g1, b1, bX1, bX1h);
    gemm_l2_k<<<gg, gb, 0, stream>>>(bX1h, wT0 + 4 * (size_t)WSZ, bm1, nullptr,
                                     nullptr, bHh, D_, D_, EP_BIAS | EP_RELU | EP_BF16);
    gemm_l2_k<<<gg, gb, 0, stream>>>(bHh, wT0 + 5 * (size_t)WSZ, bm2, nullptr,
                                     bZ, nullptr, D_, D_, EP_BIAS | EP_F32);
    res_norm_k<<<ROWS, 256, 0, stream>>>(bZ, bX1, g2, b2, bX, bXh);
  }

  tnorm_k<<<ROWS, 256, 0, stream>>>(bX, tbuf);
  pool_k<<<32, 256, 0, stream>>>(bX, tbuf, pooled_s, pooled_t);
  cls_k<<<dim3(32, 4), 256, 0, stream>>>(pooled_s, pooled_t, Wc, bc, (float*)d_out);
}
